// Model_82420422410749
// MI455X (gfx1250) — matrix operand order repaired and checked by reading, not compiled, not run on hardware
//
#include <hip/hip_runtime.h>

// ---------------- model dimensions ----------------
#define BB    32
#define SEQL  384      // encoder length == decoder length (LABEL+PRED)
#define LBL   192
#define PRD   192
#define DML   512
#define DFFL  2048
#define CIN   21
#define NMK   4
#define KTOP  17       // int(3*ln(384))
#define PHD   128

typedef __bf16 bf16;
typedef bf16  v16bf  __attribute__((ext_vector_type(16)));
typedef bf16  bf16x8 __attribute__((ext_vector_type(8)));
typedef float v8f    __attribute__((ext_vector_type(8)));
typedef float f32x4  __attribute__((ext_vector_type(4)));

static __device__ __forceinline__ bf16 f2bf(float f) {
    unsigned u = __float_as_uint(f);
    u += 0x7FFFu + ((u >> 16) & 1u);          // round-to-nearest-even
    unsigned short h = (unsigned short)(u >> 16);
    return __builtin_bit_cast(bf16, h);
}

// ---------------- CDNA5 async global->LDS copy (16B per lane) ----------------
#if __has_builtin(__builtin_amdgcn_global_load_async_to_lds_b128) && \
    __has_builtin(__builtin_amdgcn_s_wait_asynccnt)
#define HAVE_ASYNC 1
#else
#define HAVE_ASYNC 0
#endif

typedef int v4i __attribute__((ext_vector_type(4)));
typedef __attribute__((address_space(1))) v4i as1_v4i;   // global
typedef __attribute__((address_space(3))) v4i as3_v4i;   // LDS

static __device__ __forceinline__ void cp16(const bf16* g, bf16* l) {
#if HAVE_ASYNC
    __builtin_amdgcn_global_load_async_to_lds_b128((as1_v4i*)g, (as3_v4i*)l, 0, 0);
#else
    *(bf16x8*)l = *(const bf16x8*)g;
#endif
}
static __device__ __forceinline__ void async_wait() {
#if HAVE_ASYNC
    __builtin_amdgcn_s_wait_asynccnt(0);
#endif
}

// ---------------- WMMA GEMM ----------------
// C[M,N] = epilogue(A[M,K] @ B^T) ; A is [M,K] bf16 row-major, B is [N,K] bf16 row-major.
// Requires M%64==0, N%128==0, K%32==0 (true for every call site here).
#define GF_BIAS   1
#define GF_GELU   2
#define GF_RES    4
#define GF_SAVEBF 8

__global__ __launch_bounds__(256)
void k_wmma_gemm(const bf16* __restrict__ A, const bf16* __restrict__ B,
                 const float* __restrict__ bias, const float* __restrict__ res,
                 float* __restrict__ C, bf16* __restrict__ Cbf,
                 int N, int K,
                 long long sA, long long sB, long long sC, int flags)
{
    __shared__ bf16 As[2][64][40];    // [buf][m][k]  pitch 40 keeps 16B alignment
    __shared__ bf16 Bs[2][128][40];   // [buf][n][k]

    const int bz = blockIdx.z;
    A += (long long)bz * sA;
    B += (long long)bz * sB;
    C += (long long)bz * sC;
    const float* R = res ? res + (long long)bz * sC : nullptr;

    const int m0   = blockIdx.y * 64;
    const int n0   = blockIdx.x * 128;
    const int tid  = threadIdx.x;
    const int lane = tid & 31;
    const int wave = tid >> 5;
    const int wm   = wave >> 2;     // 0..1  (32 rows each)
    const int wn   = wave & 3;      // 0..3  (32 cols each)
    const int l15  = lane & 15;
    const int lhi  = lane >> 4;     // 0/1

    // staging assignments: A tile 64x32 bf16 (one 16B seg/thread),
    //                      B tile 128x32 bf16 (two 16B segs/thread)
    const int arow = tid >> 2, aseg = (tid & 3) << 3;
    const int brow = tid >> 1, bseg = (tid & 1) << 4;
    const bf16* agp = A + (long long)(m0 + arow) * K + aseg;
    const bf16* bgp = B + (long long)(n0 + brow) * K + bseg;

    auto issue = [&](int buf, int kb) {
        cp16(agp + kb, &As[buf][arow][aseg]);
        cp16(bgp + kb, &Bs[buf][brow][bseg]);
        cp16(bgp + kb + 8, &Bs[buf][brow][bseg + 8]);
    };

    v8f acc[2][2];
#pragma unroll
    for (int i = 0; i < 2; i++)
#pragma unroll
        for (int j = 0; j < 2; j++)
#pragma unroll
            for (int e = 0; e < 8; e++) acc[i][j][e] = 0.f;

    issue(0, 0);
    int cur = 0;
    for (int kb = 0; kb < K; kb += 32) {
        async_wait();
        __syncthreads();
        if (kb + 32 < K) issue(cur ^ 1, kb + 32);

        // ---- build fragments (per ISA 16-bit A/B lane layouts) ----
        v16bf af[2], bv[2];
#pragma unroll
        for (int i = 0; i < 2; i++) {
            const bf16* arp = &As[cur][wm * 32 + i * 16 + l15][0];
            bf16x8 a0 = *(const bf16x8*)(arp + 8 * lhi);
            bf16x8 a1 = *(const bf16x8*)(arp + 8 * lhi + 16);
#pragma unroll
            for (int e = 0; e < 8; e++) { af[i][e] = a0[e]; af[i][e + 8] = a1[e]; }
        }
#pragma unroll
        for (int j = 0; j < 2; j++) {
            const bf16* brp = &Bs[cur][wn * 32 + j * 16 + l15][0];
            bf16x8 b0 = *(const bf16x8*)(brp + 8 * lhi);
            bf16x8 b1 = *(const bf16x8*)(brp + 8 * lhi + 16);
#pragma unroll
            for (int e = 0; e < 8; e++) { bv[j][e] = b0[e]; bv[j][e + 8] = b1[e]; }
        }
#pragma unroll
        for (int i = 0; i < 2; i++)
#pragma unroll
            for (int j = 0; j < 2; j++)
                acc[i][j] = __builtin_amdgcn_wmma_f32_16x16x32_bf16(
                    false, af[i], false, bv[j], (short)0, acc[i][j], false, false);

        cur ^= 1;
    }

    // ---- epilogue: bias / exact GELU / residual (+ optional bf16 mirror) ----
#pragma unroll
    for (int i = 0; i < 2; i++) {
        int rbase = m0 + wm * 32 + i * 16 + lhi * 8;
#pragma unroll
        for (int j = 0; j < 2; j++) {
            int col = n0 + wn * 32 + j * 16 + l15;
            float bvv = (flags & GF_BIAS) ? bias[col] : 0.f;
#pragma unroll
            for (int r = 0; r < 8; r++) {
                float v = acc[i][j][r] + bvv;
                if (flags & GF_GELU) v = 0.5f * v * (1.f + erff(v * 0.70710678118654752f));
                long long off = (long long)(rbase + r) * N + col;
                if (flags & GF_RES) v += R[off];
                C[off] = v;
                if (flags & GF_SAVEBF) Cbf[off] = f2bf(v);   // non-batched call sites only
            }
        }
    }
}

// weight converter: W [K,N] fp32 row-major  ->  Wt [N,K] bf16 row-major
__global__ void k_cvt_wt(const float* __restrict__ W, bf16* __restrict__ Wt, int K, int N)
{
    long long i = (long long)blockIdx.x * blockDim.x + threadIdx.x;
    if (i >= (long long)K * N) return;
    int n = (int)(i % N);
    int k = (int)(i / N);
    Wt[(long long)n * K + k] = f2bf(W[i]);
}

// ---------------- small kernels ----------------
__global__ void k_stats(const float* __restrict__ x, float* __restrict__ mean, float* __restrict__ stdv)
{
    int i = blockIdx.x * blockDim.x + threadIdx.x;
    if (i >= BB * CIN) return;
    int b = i / CIN, c = i % CIN;
    const float* p = x + (long long)b * SEQL * CIN + c;
    float s = 0.f;
    for (int t = 0; t < SEQL; t++) s += p[(long long)t * CIN];
    float m = s / SEQL;
    float v = 0.f;
    for (int t = 0; t < SEQL; t++) { float d = p[(long long)t * CIN] - m; v += d * d; }
    mean[i] = m;
    stdv[i] = sqrtf(v / SEQL + 1e-5f);
}

__global__ void k_normalize(const float* __restrict__ x, const float* __restrict__ mean,
                            const float* __restrict__ stdv, float* __restrict__ y)
{
    int i = blockIdx.x * blockDim.x + threadIdx.x;
    if (i >= BB * SEQL * CIN) return;
    int c = i % CIN;
    int b = i / (SEQL * CIN);
    y[i] = (x[i] - mean[b * CIN + c]) / stdv[b * CIN + c];
}

// tau/delta projector: channel-circular conv over SEQ taps + 42->128->128->out MLP
__global__ __launch_bounds__(128)
void k_projector(const float* __restrict__ xraw, const float* __restrict__ stats,
                 const float* __restrict__ conv,
                 const float* __restrict__ W0, const float* __restrict__ b0,
                 const float* __restrict__ W1, const float* __restrict__ b1,
                 const float* __restrict__ W2,
                 float* __restrict__ out, int out_dim, int do_exp)
{
    __shared__ float h0[2 * CIN];
    __shared__ float h1[PHD];
    __shared__ float h2[PHD];
    int b = blockIdx.x, tid = threadIdx.x;
    if (tid < CIN) {
        float s = 0.f;
        for (int t = 0; t < SEQL; t++) {
            const float* xr = xraw + ((long long)b * SEQL + t) * CIN;
            const float* cv = conv + t * 3;
            for (int j = 0; j < 3; j++) {
                int cc  = tid + j;                              // padded channel index 0..22
                int src = (cc == 0) ? CIN - 1 : (cc == CIN + 1) ? 0 : cc - 1;
                s += xr[src] * cv[j];
            }
        }
        h0[tid]       = s;
        h0[CIN + tid] = stats[b * CIN + tid];
    }
    __syncthreads();
    if (tid < PHD) {
        float s = b0[tid];
        for (int i = 0; i < 2 * CIN; i++) s += h0[i] * W0[i * PHD + tid];
        h1[tid] = s > 0.f ? s : 0.f;
    }
    __syncthreads();
    if (tid < PHD) {
        float s = b1[tid];
        for (int i = 0; i < PHD; i++) s += h1[i] * W1[i * PHD + tid];
        h2[tid] = s > 0.f ? s : 0.f;
    }
    __syncthreads();
    for (int o = tid; o < out_dim; o += blockDim.x) {
        float s = 0.f;
        for (int i = 0; i < PHD; i++) s += h2[i] * W2[i * out_dim + o];
        out[b * out_dim + o] = do_exp ? expf(s) : s;
    }
}

// moving-average (k=25, edge replicate) decomposition; optional bf16 mirror of seasonal
__global__ void k_decomp(const float* __restrict__ x, float* __restrict__ seasonal,
                         bf16* __restrict__ seasonal_bf, float* __restrict__ trend,
                         int L, int D, int acc_trend)
{
    long long i = (long long)blockIdx.x * blockDim.x + threadIdx.x;
    long long total = (long long)BB * L * D;
    if (i >= total) return;
    int d = (int)(i % D);
    int t = (int)((i / D) % L);
    int b = (int)(i / ((long long)L * D));
    const float* row = x + (long long)b * L * D + d;
    float s = 0.f;
    for (int j = t - 12; j <= t + 12; j++) {
        int tc = j < 0 ? 0 : (j >= L ? L - 1 : j);
        s += row[(long long)tc * D];
    }
    s *= (1.f / 25.f);
    if (trend)    { if (acc_trend) trend[i] += s; else trend[i] = s; }
    if (seasonal) {
        float sv = x[i] - s;
        seasonal[i] = sv;
        if (seasonal_bf) seasonal_bf[i] = f2bf(sv);
    }
}

__global__ void k_time_mean(const float* __restrict__ x, float* __restrict__ out, int L, int D)
{
    int i = blockIdx.x * blockDim.x + threadIdx.x;
    if (i >= BB * D) return;
    int b = i / D, d = i % D;
    const float* p = x + (long long)b * L * D + d;
    float s = 0.f;
    for (int t = 0; t < L; t++) s += p[(long long)t * D];
    out[i] = s / L;
}

__global__ void k_sub_cols(float* __restrict__ x, const float* __restrict__ cm,
                           bf16* __restrict__ xbf, long long total, int D)
{
    long long i = (long long)blockIdx.x * blockDim.x + threadIdx.x;
    if (i >= total) return;
    int d = (int)(i % D);
    int b = (int)(i / ((long long)SEQL * D));
    float v = x[i] - cm[b * D + d];
    x[i] = v;
    if (xbf) xbf[i] = f2bf(v);
}

__global__ void k_dec_init(const float* __restrict__ seas_xn, const float* __restrict__ trend_xn,
                           const float* __restrict__ meanT,
                           float* __restrict__ seas_init, float* __restrict__ trend_init)
{
    int i = blockIdx.x * blockDim.x + threadIdx.x;
    if (i >= BB * SEQL * CIN) return;
    int c = i % CIN;
    int t = (i / CIN) % SEQL;
    int b = i / (SEQL * CIN);
    if (t < LBL) {
        int ts = SEQL - LBL + t;
        seas_init[i]  = seas_xn[((long long)b * SEQL + ts) * CIN + c];
        trend_init[i] = trend_xn[((long long)b * SEQL + ts) * CIN + c];
    } else {
        seas_init[i]  = 0.f;
        trend_init[i] = meanT[b * CIN + c];
    }
}

// circular-time conv3 embedding + time-mark linear embedding (fp32 + bf16 mirror)
__global__ void k_embed(const float* __restrict__ x, const float* __restrict__ w,
                        const float* __restrict__ mark, const float* __restrict__ emb,
                        float* __restrict__ out, bf16* __restrict__ outbf)
{
    long long i = (long long)blockIdx.x * blockDim.x + threadIdx.x;
    if (i >= (long long)BB * SEQL * DML) return;
    int d = (int)(i % DML);
    int t = (int)((i / DML) % SEQL);
    int b = (int)(i / ((long long)SEQL * DML));
    const float* xb = x + (long long)b * SEQL * CIN;
    float s = 0.f;
    for (int j = 0; j < 3; j++) {
        int tt = t - 1 + j;
        tt = tt < 0 ? SEQL - 1 : (tt >= SEQL ? 0 : tt);
        const float* xr = xb + (long long)tt * CIN;
        const float* wr = w + d * CIN * 3 + j;
        for (int c = 0; c < CIN; c++) s += xr[c] * wr[c * 3];
    }
    const float* mk = mark + ((long long)b * SEQL + t) * NMK;
    for (int m = 0; m < NMK; m++) s += mk[m] * emb[m * DML + d];
    out[i]   = s;
    outbf[i] = f2bf(s);
}

// mean_corr[b,tau] = tau_s[b]/DM * sum_t S[b,t,(t+tau)%L]  (+ delta)
__global__ void k_corr_mean(const float* __restrict__ S, const float* __restrict__ tau_s,
                            const float* __restrict__ delta, float* __restrict__ mc)
{
    int i = blockIdx.x * blockDim.x + threadIdx.x;
    if (i >= BB * SEQL) return;
    int b  = i / SEQL;
    int tp = i % SEQL;
    const float* Sb = S + (long long)b * SEQL * SEQL;
    float s = 0.f;
    for (int t = 0; t < SEQL; t++) {
        int tt = t + tp; if (tt >= SEQL) tt -= SEQL;
        s += Sb[(long long)t * SEQL + tt];
    }
    s = s * tau_s[b] * (1.f / DML);
    if (delta) s += delta[b * SEQL + tp];
    mc[i] = s;
}

__global__ void k_topk(const float* __restrict__ mc, int* __restrict__ delay, float* __restrict__ wts)
{
    __shared__ float sm[SEQL];
    int b = blockIdx.x;
    for (int t = threadIdx.x; t < SEQL; t += blockDim.x) sm[t] = mc[b * SEQL + t];
    __syncthreads();
    if (threadIdx.x == 0) {
        float w[KTOP];
        for (int i = 0; i < KTOP; i++) {
            int am = 0; float mv = -1e30f;
            for (int t = 0; t < SEQL; t++) if (sm[t] > mv) { mv = sm[t]; am = t; }
            w[i] = mv; delay[b * KTOP + i] = am; sm[am] = -1e30f;
        }
        float mx = w[0];
        for (int i = 1; i < KTOP; i++) if (w[i] > mx) mx = w[i];
        float sum = 0.f;
        for (int i = 0; i < KTOP; i++) { w[i] = expf(w[i] - mx); sum += w[i]; }
        float inv = 1.f / sum;
        for (int i = 0; i < KTOP; i++) wts[b * KTOP + i] = w[i] * inv;
    }
}

__global__ __launch_bounds__(256)
void k_delay_agg(const float* __restrict__ V, const int* __restrict__ delay,
                 const float* __restrict__ wts, float* __restrict__ out, bf16* __restrict__ outbf)
{
    __shared__ int   sd[KTOP];
    __shared__ float sw[KTOP];
    int b = blockIdx.z, t = blockIdx.y;
    int d = blockIdx.x * blockDim.x + threadIdx.x;
    if (threadIdx.x < KTOP) {
        sd[threadIdx.x] = delay[b * KTOP + threadIdx.x];
        sw[threadIdx.x] = wts[b * KTOP + threadIdx.x];
    }
    __syncthreads();
    const float* Vb = V + (long long)b * SEQL * DML;
    float s = 0.f;
    for (int i = 0; i < KTOP; i++) {
        int tt = t + sd[i]; if (tt >= SEQL) tt -= SEQL;
        s += sw[i] * Vb[(long long)tt * DML + d];
    }
    long long o = ((long long)b * SEQL + t) * DML + d;
    out[o]   = s;
    outbf[o] = f2bf(s);
}

__global__ __launch_bounds__(256)
void k_ln(const float* __restrict__ x, const float* __restrict__ g, const float* __restrict__ be,
          float* __restrict__ y)
{
    __shared__ float red[256];
    long long row = blockIdx.x;
    const float* xr = x + row * DML;
    int tid = threadIdx.x;
    float s = 0.f;
    for (int d = tid; d < DML; d += 256) s += xr[d];
    red[tid] = s; __syncthreads();
    for (int o = 128; o > 0; o >>= 1) { if (tid < o) red[tid] += red[tid + o]; __syncthreads(); }
    float mu = red[0] * (1.f / DML);
    __syncthreads();
    float v = 0.f;
    for (int d = tid; d < DML; d += 256) { float t = xr[d] - mu; v += t * t; }
    red[tid] = v; __syncthreads();
    for (int o = 128; o > 0; o >>= 1) { if (tid < o) red[tid] += red[tid + o]; __syncthreads(); }
    float inv = rsqrtf(red[0] * (1.f / DML) + 1e-5f);
    for (int d = tid; d < DML; d += 256) y[row * DML + d] = (xr[d] - mu) * inv * g[d] + be[d];
}

// trend conv: rt[b,t,o] += sum_{d,j} tsum[b,(t-1+j)%L,d] * w[o,d,j]
__global__ void k_trend_conv(const float* __restrict__ ts, const float* __restrict__ w,
                             float* __restrict__ trend)
{
    long long i = (long long)blockIdx.x * blockDim.x + threadIdx.x;
    if (i >= (long long)BB * SEQL * CIN) return;
    int o = (int)(i % CIN);
    int t = (int)((i / CIN) % SEQL);
    int b = (int)(i / ((long long)SEQL * CIN));
    const float* tb = ts + (long long)b * SEQL * DML;
    float s = 0.f;
    for (int j = 0; j < 3; j++) {
        int tt = t - 1 + j;
        tt = tt < 0 ? SEQL - 1 : (tt >= SEQL ? 0 : tt);
        const float* xr = tb + (long long)tt * DML;
        const float* wr = w + o * DML * 3 + j;
        for (int d = 0; d < DML; d++) s += xr[d] * wr[d * 3];
    }
    trend[i] += s;
}

__global__ void k_final(const float* __restrict__ dec, const float* __restrict__ projW,
                        const float* __restrict__ projb, const float* __restrict__ trend,
                        const float* __restrict__ stdv, const float* __restrict__ mean,
                        float* __restrict__ out)
{
    long long i = (long long)blockIdx.x * blockDim.x + threadIdx.x;
    if (i >= (long long)BB * SEQL * CIN) return;
    int c = (int)(i % CIN);
    int t = (int)((i / CIN) % SEQL);
    int b = (int)(i / ((long long)SEQL * CIN));
    if (t < LBL) return;
    const float* dr = dec + ((long long)b * SEQL + t) * DML;
    float s = projb[c];
    for (int d = 0; d < DML; d++) s += dr[d] * projW[d * CIN + c];
    float v = (trend[i] + s) * stdv[b * CIN + c] + mean[b * CIN + c];
    out[((long long)b * PRD + (t - LBL)) * CIN + c] = v;
}

// ---------------- host orchestration ----------------
extern "C" void kernel_launch(void* const* d_in, const int* in_sizes, int n_in,
                              void* d_out, int out_size, void* d_ws, size_t ws_size,
                              hipStream_t stream)
{
    const float* x_enc    = (const float*)d_in[0];
    const float* mark_enc = (const float*)d_in[1];
    const float* mark_dec = (const float*)d_in[3];

    // params flattened in jax-pytree (sorted dict key) order
    const float* dec_emb_conv = (const float*)d_in[4];
    const float* dec_emb_time = (const float*)d_in[5];
    const float* dec_c1       = (const float*)d_in[6];
    const float* dec_c2       = (const float*)d_in[7];
    const float* cr_Wk = (const float*)d_in[8];
    const float* cr_Wo = (const float*)d_in[9];
    const float* cr_Wq = (const float*)d_in[10];
    const float* cr_Wv = (const float*)d_in[11];
    const float* cr_bk = (const float*)d_in[12];
    const float* cr_bo = (const float*)d_in[13];
    const float* cr_bq = (const float*)d_in[14];
    const float* cr_bv = (const float*)d_in[15];
    const float* sf_Wk = (const float*)d_in[16];
    const float* sf_Wo = (const float*)d_in[17];
    const float* sf_Wq = (const float*)d_in[18];
    const float* sf_Wv = (const float*)d_in[19];
    const float* sf_bk = (const float*)d_in[20];
    const float* sf_bo = (const float*)d_in[21];
    const float* sf_bq = (const float*)d_in[22];
    const float* sf_bv = (const float*)d_in[23];
    const float* dec_trend_w  = (const float*)d_in[24];
    const float* dec_norm_b   = (const float*)d_in[25];
    const float* dec_norm_w   = (const float*)d_in[26];
    const float* del_b0 = (const float*)d_in[27];
    const float* del_b1 = (const float*)d_in[28];
    const float* del_cv = (const float*)d_in[29];
    const float* del_W0 = (const float*)d_in[30];
    const float* del_W1 = (const float*)d_in[31];
    const float* del_W2 = (const float*)d_in[32];
    const float* enc_emb_conv = (const float*)d_in[33];
    const float* enc_emb_time = (const float*)d_in[34];
    // enc layers: base 35 + 10*L : Wk,Wo,Wq,Wv,bk,bo,bq,bv,c1,c2
    const float* enc_norm_b = (const float*)d_in[55];
    const float* enc_norm_w = (const float*)d_in[56];
    const float* proj_W     = (const float*)d_in[57];
    const float* proj_b     = (const float*)d_in[58];
    const float* tau_b0 = (const float*)d_in[59];
    const float* tau_b1 = (const float*)d_in[60];
    const float* tau_cv = (const float*)d_in[61];
    const float* tau_W0 = (const float*)d_in[62];
    const float* tau_W1 = (const float*)d_in[63];
    const float* tau_W2 = (const float*)d_in[64];

    // ---- workspace layout ----
    char* wsb = (char*)d_ws;
    size_t off = 0;
    auto alloc = [&](size_t bytes) -> float* {
        float* p = (float*)(wsb + off);
        off = (off + bytes + 255) & ~(size_t)255;
        return p;
    };
    auto allocbf = [&](size_t elems) -> bf16* { return (bf16*)alloc(elems * 2); };

    const size_t BLDM  = (size_t)BB * SEQL * DML * 4;
    const size_t NBLDM = (size_t)BB * SEQL * DML;
    float* f_xn    = alloc((size_t)BB * SEQL * CIN * 4);
    float* f_mean  = alloc((size_t)BB * CIN * 4);
    float* f_std   = alloc((size_t)BB * CIN * 4);
    float* f_meanT = alloc((size_t)BB * CIN * 4);
    float* f_tau   = alloc((size_t)BB * 4);
    float* f_delta = alloc((size_t)BB * SEQL * 4);
    float* f_sxn   = alloc((size_t)BB * SEQL * CIN * 4);
    float* f_txn   = alloc((size_t)BB * SEQL * CIN * 4);
    float* f_sinit = alloc((size_t)BB * SEQL * CIN * 4);
    float* f_trend = alloc((size_t)BB * SEQL * CIN * 4);
    float* f_enc   = alloc(BLDM);
    float* f_x     = alloc(BLDM);
    float* f_q     = alloc(BLDM);
    float* f_k     = alloc(BLDM);
    float* f_v     = alloc(BLDM);
    float* f_y     = alloc(BLDM);
    float* f_t2    = alloc(BLDM);
    float* f_t3    = alloc(BLDM);
    float* f_tsum  = alloc(BLDM);
    float* f_S     = alloc((size_t)BB * SEQL * SEQL * 4);
    float* f_mc    = alloc((size_t)BB * SEQL * 4);
    int*   i_delay = (int*)alloc((size_t)BB * KTOP * 4);
    float* f_wts   = alloc((size_t)BB * KTOP * 4);
    float* f_cm    = alloc((size_t)BB * DML * 4);
    float* f_h     = alloc((size_t)BB * SEQL * DFFL * 4);
    // bf16 activation mirrors
    bf16* b_enc = allocbf(NBLDM);
    bf16* b_x   = allocbf(NBLDM);
    bf16* b_q   = allocbf(NBLDM);
    bf16* b_k   = allocbf(NBLDM);
    bf16* b_y   = allocbf(NBLDM);
    bf16* b_t2  = allocbf(NBLDM);
    bf16* b_t3  = allocbf(NBLDM);
    bf16* b_h   = allocbf((size_t)BB * SEQL * DFFL);
    // bf16 transposed [N,K] weight mirrors
    bf16* t_sfW[4]; bf16* t_crW[4]; bf16* t_eW[2][4];
    for (int i = 0; i < 4; i++) { t_sfW[i] = allocbf(DML * DML); t_crW[i] = allocbf(DML * DML); }
    for (int l = 0; l < 2; l++) for (int i = 0; i < 4; i++) t_eW[l][i] = allocbf(DML * DML);
    bf16* t_dc1 = allocbf((size_t)DML * DFFL);
    bf16* t_dc2 = allocbf((size_t)DML * DFFL);
    bf16* t_ec1[2]; bf16* t_ec2[2];
    for (int l = 0; l < 2; l++) { t_ec1[l] = allocbf((size_t)DML * DFFL); t_ec2[l] = allocbf((size_t)DML * DFFL); }
    (void)ws_size; (void)n_in; (void)in_sizes; (void)out_size;

    auto grid1 = [](long long total) { return (unsigned)((total + 255) / 256); };
    auto cvt = [&](const float* W, bf16* Wt, int Kd, int Nd) {
        k_cvt_wt<<<grid1((long long)Kd * Nd), 256, 0, stream>>>(W, Wt, Kd, Nd);
    };
    // convert all GEMM weights once (order: q,k,v,o)
    cvt(sf_Wq, t_sfW[0], DML, DML); cvt(sf_Wk, t_sfW[1], DML, DML);
    cvt(sf_Wv, t_sfW[2], DML, DML); cvt(sf_Wo, t_sfW[3], DML, DML);
    cvt(cr_Wq, t_crW[0], DML, DML); cvt(cr_Wk, t_crW[1], DML, DML);
    cvt(cr_Wv, t_crW[2], DML, DML); cvt(cr_Wo, t_crW[3], DML, DML);
    for (int l = 0; l < 2; l++) {
        int base = 35 + 10 * l;
        cvt((const float*)d_in[base + 2], t_eW[l][0], DML, DML);   // Wq
        cvt((const float*)d_in[base + 0], t_eW[l][1], DML, DML);   // Wk
        cvt((const float*)d_in[base + 3], t_eW[l][2], DML, DML);   // Wv
        cvt((const float*)d_in[base + 1], t_eW[l][3], DML, DML);   // Wo
        cvt((const float*)d_in[base + 8], t_ec1[l], DML, DFFL);
        cvt((const float*)d_in[base + 9], t_ec2[l], DFFL, DML);
    }
    cvt(dec_c1, t_dc1, DML, DFFL);
    cvt(dec_c2, t_dc2, DFFL, DML);

    const int M = BB * SEQL;
    auto gemm = [&](const bf16* A, const bf16* Bw, const float* bias, const float* res,
                    float* C, bf16* Cbf, int m, int n, int kdim, int batch,
                    long long sA, long long sB, long long sC, int flags) {
        dim3 g(n / 128, m / 64, batch);
        k_wmma_gemm<<<g, dim3(256), 0, stream>>>(A, Bw, bias, res, C, Cbf, n, kdim, sA, sB, sC, flags);
    };
    auto attention = [&](const bf16* xq_bf, const float* xq_res, const bf16* xkv_bf,
                         bf16* const* Wt, const float* bq, const float* bk, const float* bv,
                         const float* bo, const float* delta, float* outY) {
        gemm(xq_bf,  Wt[0], bq, nullptr, f_q, b_q, M, DML, DML, 1, 0, 0, 0, GF_BIAS | GF_SAVEBF);
        gemm(xkv_bf, Wt[1], bk, nullptr, f_k, b_k, M, DML, DML, 1, 0, 0, 0, GF_BIAS | GF_SAVEBF);
        gemm(xkv_bf, Wt[2], bv, nullptr, f_v, nullptr, M, DML, DML, 1, 0, 0, 0, GF_BIAS);
        // S[b] = Q[b] @ K[b]^T   (K's natural [t,d] layout == required [N,K])
        gemm(b_q, b_k, nullptr, nullptr, f_S, nullptr, SEQL, SEQL, DML, BB,
             (long long)SEQL * DML, (long long)SEQL * DML, (long long)SEQL * SEQL, 0);
        k_corr_mean<<<(BB * SEQL + 255) / 256, 256, 0, stream>>>(f_S, f_tau, delta, f_mc);
        k_topk<<<BB, 128, 0, stream>>>(f_mc, i_delay, f_wts);
        k_delay_agg<<<dim3(DML / 256, SEQL, BB), 256, 0, stream>>>(f_v, i_delay, f_wts, f_y, b_y);
        gemm(b_y, Wt[3], bo, xq_res, outY, nullptr, M, DML, DML, 1, 0, 0, 0, GF_BIAS | GF_RES);
    };
    auto ffn = [&](const bf16* xin_bf, const float* xin_res, const bf16* c1t, const bf16* c2t,
                   float* outY) {
        gemm(xin_bf, c1t, nullptr, nullptr, f_h, b_h, M, DFFL, DML, 1, 0, 0, 0, GF_GELU | GF_SAVEBF);
        gemm(b_h, c2t, nullptr, xin_res, outY, nullptr, M, DML, DFFL, 1, 0, 0, 0, GF_RES);
    };

    // ---- stats, normalize, projectors ----
    k_stats<<<(BB * CIN + 127) / 128, 128, 0, stream>>>(x_enc, f_mean, f_std);
    k_normalize<<<grid1((long long)BB * SEQL * CIN), 256, 0, stream>>>(x_enc, f_mean, f_std, f_xn);
    k_projector<<<BB, 128, 0, stream>>>(x_enc, f_std, tau_cv, tau_W0, tau_b0, tau_W1, tau_b1,
                                        tau_W2, f_tau, 1, 1);
    k_projector<<<BB, 128, 0, stream>>>(x_enc, f_mean, del_cv, del_W0, del_b0, del_W1, del_b1,
                                        del_W2, f_delta, SEQL, 0);
    // ---- decoder init (decomp of normalized input) ----
    k_decomp<<<grid1((long long)BB * SEQL * CIN), 256, 0, stream>>>(f_xn, f_sxn, nullptr, f_txn,
                                                                    SEQL, CIN, 0);
    k_time_mean<<<(BB * CIN + 127) / 128, 128, 0, stream>>>(f_xn, f_meanT, SEQL, CIN);
    k_dec_init<<<grid1((long long)BB * SEQL * CIN), 256, 0, stream>>>(f_sxn, f_txn, f_meanT,
                                                                      f_sinit, f_trend);
    // ---- encoder ----
    k_embed<<<grid1((long long)BB * SEQL * DML), 256, 0, stream>>>(f_xn, enc_emb_conv, mark_enc,
                                                                   enc_emb_time, f_enc, b_enc);
    for (int l = 0; l < 2; l++) {
        int base = 35 + 10 * l;
        const float* bk = (const float*)d_in[base + 4];
        const float* bo = (const float*)d_in[base + 5];
        const float* bq = (const float*)d_in[base + 6];
        const float* bv = (const float*)d_in[base + 7];
        attention(b_enc, f_enc, b_enc, t_eW[l], bq, bk, bv, bo, f_delta, f_x);
        k_decomp<<<grid1((long long)NBLDM), 256, 0, stream>>>(f_x, f_t2, b_t2, nullptr, SEQL, DML, 0);
        ffn(b_t2, f_t2, t_ec1[l], t_ec2[l], f_x);
        k_decomp<<<grid1((long long)NBLDM), 256, 0, stream>>>(f_x, f_enc, b_enc, nullptr, SEQL, DML, 0);
    }
    // enc my_ln -> f_x (enc_out) with bf16 mirror b_x
    k_ln<<<BB * SEQL, 256, 0, stream>>>(f_enc, enc_norm_w, enc_norm_b, f_x);
    k_time_mean<<<(BB * DML + 255) / 256, 256, 0, stream>>>(f_x, f_cm, SEQL, DML);
    k_sub_cols<<<grid1((long long)NBLDM), 256, 0, stream>>>(f_x, f_cm, b_x, (long long)NBLDM, DML);
    // ---- decoder ----
    float* dx = f_enc;
    k_embed<<<grid1((long long)BB * SEQL * DML), 256, 0, stream>>>(f_sinit, dec_emb_conv, mark_dec,
                                                                   dec_emb_time, dx, b_enc);
    // self attention (delta = None)
    attention(b_enc, dx, b_enc, t_sfW, sf_bq, sf_bk, sf_bv, sf_bo, nullptr, f_t2);
    k_decomp<<<grid1((long long)NBLDM), 256, 0, stream>>>(f_t2, f_t3, b_t3, f_tsum, SEQL, DML, 0);
    // cross attention (delta = None at decoder per reference call)
    attention(b_t3, f_t3, b_x, t_crW, cr_bq, cr_bk, cr_bv, cr_bo, nullptr, f_t2);
    k_decomp<<<grid1((long long)NBLDM), 256, 0, stream>>>(f_t2, dx, b_enc, f_tsum, SEQL, DML, 1);
    ffn(b_enc, dx, t_dc1, t_dc2, f_t2);
    k_decomp<<<grid1((long long)NBLDM), 256, 0, stream>>>(f_t2, f_t3, nullptr, f_tsum, SEQL, DML, 1);
    k_trend_conv<<<grid1((long long)BB * SEQL * CIN), 256, 0, stream>>>(f_tsum, dec_trend_w, f_trend);
    // dec my_ln -> f_t2
    k_ln<<<BB * SEQL, 256, 0, stream>>>(f_t3, dec_norm_w, dec_norm_b, f_t2);
    k_time_mean<<<(BB * DML + 255) / 256, 256, 0, stream>>>(f_t2, f_cm, SEQL, DML);
    k_sub_cols<<<grid1((long long)NBLDM), 256, 0, stream>>>(f_t2, f_cm, nullptr, (long long)NBLDM, DML);
    // final projection + de-normalization, slice last PRED
    k_final<<<grid1((long long)BB * SEQL * CIN), 256, 0, stream>>>(f_t2, proj_W, proj_b, f_trend,
                                                                   f_std, f_mean, (float*)d_out);
}
